// LSTM_40673340293447
// MI455X (gfx1250) — matrix operand order repaired and checked by reading, not compiled, not run on hardware
//
#include <hip/hip_runtime.h>

// ---------- types ----------
typedef __attribute__((ext_vector_type(16))) __bf16 v16bf;
typedef __attribute__((ext_vector_type(8)))  __bf16 v8bf;
typedef __attribute__((ext_vector_type(8)))  float  v8f;

static __device__ __forceinline__ unsigned short f2bf_rne(float f) {
    unsigned int u = __float_as_uint(f);
    unsigned int r = (u + 0x7FFFu + ((u >> 16) & 1u)) >> 16;
    return (unsigned short)r;
}
static __device__ __forceinline__ float bf2f(unsigned short u) {
    return __uint_as_float((unsigned int)u << 16);
}

// ---------- prep kernels ----------
__global__ void k_f2bf(const float* __restrict__ src, unsigned short* __restrict__ dst, int n) {
    int i = blockIdx.x * 256 + threadIdx.x;
    if (i < n) dst[i] = f2bf_rne(src[i]);
}
__global__ void k_copyf(const float* __restrict__ src, float* __restrict__ dst, int n) {
    int i = blockIdx.x * 256 + threadIdx.x;
    if (i < n) dst[i] = src[i];
}
__global__ void k_copybf(const unsigned short* __restrict__ src, unsigned short* __restrict__ dst, int n) {
    int i = blockIdx.x * 256 + threadIdx.x;
    if (i < n) dst[i] = src[i];
}

// ---------- parallel input GEMM: Zx[M,4096] = X[M,K1] @ W[:, 0:K1]^T (bf16, f32 acc) ----------
// M = 2048 (one 64-timestep chunk). grid = 4096 blocks x 128 thr; wave = one 32(M)x16(N) tile.
__global__ __launch_bounds__(128)
void gemm_zx(const unsigned short* __restrict__ X, int K1,      // [M, K1] bf16
             const unsigned short* __restrict__ W, int K,       // [4096, K] bf16, cols 0..K1 used
             unsigned short* __restrict__ Zx)                   // [M, 4096] bf16
{
    const int tid  = threadIdx.x;
    const int wave = tid >> 5;
    const int lane = tid & 31;
    const int w    = blockIdx.x * 4 + wave;     // 0..16383
    const int mpair = w >> 8;                   // 0..63  -> rows mpair*32..+31
    const int ntile = w & 255;                  // 0..255 -> cols ntile*16..+15
    const int mbase = mpair * 32;
    const int ncol  = ntile * 16;

    const int mlo   = lane & 15;
    const int khalf = lane >> 4;
    const int kbA   = khalf * 8;
    const int kbB   = khalf * 16;
    const int nB    = lane & 15;

    v8f acc0 = {}, acc1 = {};
    const unsigned short* Wrow = W + (size_t)(ncol + nB) * (size_t)K;

    for (int kt = 0; kt < (K1 >> 5); ++kt) {
        const int k0 = kt << 5;
        v8bf blo = *(const v8bf*)(Wrow + k0 + kbA);
        v8bf bhi = *(const v8bf*)(Wrow + k0 + kbA + 16);
        v16bf bfrag = __builtin_shufflevector(blo, bhi, 0,1,2,3,4,5,6,7,8,9,10,11,12,13,14,15);

        const unsigned short* arow0 = X + (size_t)(mbase + mlo) * K1 + k0 + kbA;
        v8bf a0lo = *(const v8bf*)(arow0);
        v8bf a0hi = *(const v8bf*)(arow0 + 16);
        v16bf afrag0 = __builtin_shufflevector(a0lo, a0hi, 0,1,2,3,4,5,6,7,8,9,10,11,12,13,14,15);

        const unsigned short* arow1 = arow0 + (size_t)16 * K1;
        v8bf a1lo = *(const v8bf*)(arow1);
        v8bf a1hi = *(const v8bf*)(arow1 + 16);
        v16bf afrag1 = __builtin_shufflevector(a1lo, a1hi, 0,1,2,3,4,5,6,7,8,9,10,11,12,13,14,15);

        acc0 = __builtin_amdgcn_wmma_f32_16x16x32_bf16(false, afrag0, false, bfrag,
                                                       (short)0, acc0, false, false);
        acc1 = __builtin_amdgcn_wmma_f32_16x16x32_bf16(false, afrag1, false, bfrag,
                                                       (short)0, acc1, false, false);
    }

#pragma unroll
    for (int r = 0; r < 8; ++r) {
        const int m0 = mbase + r + khalf * 8;
        Zx[(size_t)m0 * 4096 + ncol + nB]        = f2bf_rne(acc0[r]);
        Zx[(size_t)(m0 + 16) * 4096 + ncol + nB] = f2bf_rne(acc1[r]);
    }
}

// ---------- fused LSTM step ----------
// z = (zx? Zx : inp-part GEMM) + hprev @ Wh^T + b ; gates ; cell update.
// grid = (64 n-tiles, 2 m-tiles) x 128 thr; wave g = gate g's 16x16 z tile (one WMMA acc).
__global__ __launch_bounds__(128)
void lstm_step(const unsigned short* __restrict__ inp, int K1,   // [32,K1] bf16 (unused if zx)
               const unsigned short* __restrict__ zx,            // [32,4096] bf16 or null
               const unsigned short* __restrict__ hprev,         // [32,1024] bf16
               const unsigned short* __restrict__ W, int K,      // [4096,K] bf16
               const float* __restrict__ bias,                   // [4096] f32
               float* __restrict__ c,                            // [32,1024] f32 state
               unsigned short* __restrict__ hnext,               // [32,1024] bf16
               float* __restrict__ out)                          // [32,1024] f32 or null
{
    __shared__ float zbuf[4][16][16];

    const int tid   = threadIdx.x;
    const int wave  = tid >> 5;          // gate 0..3
    const int lane  = tid & 31;
    const int mtile = blockIdx.y;        // 0..1 (batch half)
    const int ncol0 = blockIdx.x * 16;   // 0..1008

    const int mlo   = lane & 15;
    const int khalf = lane >> 4;
    const int kbA   = khalf * 8;
    const int kbB   = khalf * 16;
    const int nB    = lane & 15;

    v8f acc = {};
    const unsigned short* Wrow = W + (size_t)(wave * 1024 + ncol0 + nB) * (size_t)K;

    const int k1iters = K1 >> 5;
    const int kstart  = zx ? k1iters : 0;

    for (int kt = kstart; kt < (K >> 5); ++kt) {
        const int k0 = kt << 5;
        const unsigned short* src;
        int stride, koff;
        if (kt < k1iters) { src = inp;   stride = K1;   koff = k0; }
        else              { src = hprev; stride = 1024; koff = k0 - K1; }

        v8bf blo = *(const v8bf*)(Wrow + k0 + kbA);
        v8bf bhi = *(const v8bf*)(Wrow + k0 + kbA + 16);
        v16bf bfrag = __builtin_shufflevector(blo, bhi, 0,1,2,3,4,5,6,7,8,9,10,11,12,13,14,15);

        const unsigned short* arow = src + (size_t)(mtile * 16 + mlo) * stride + koff + kbA;
        v8bf alo = *(const v8bf*)(arow);
        v8bf ahi = *(const v8bf*)(arow + 16);
        v16bf afrag = __builtin_shufflevector(alo, ahi, 0,1,2,3,4,5,6,7,8,9,10,11,12,13,14,15);

        acc = __builtin_amdgcn_wmma_f32_16x16x32_bf16(false, afrag, false, bfrag,
                                                      (short)0, acc, false, false);
    }

#pragma unroll
    for (int r = 0; r < 8; ++r)
        zbuf[wave][r + khalf * 8][nB] = acc[r];
    __syncthreads();

    // 16 batch rows x 16 cols = 256 outputs, 2 per thread
#pragma unroll
    for (int i = 0; i < 2; ++i) {
        const int idx  = tid + i * 128;      // 0..255
        const int brow = idx >> 4;
        const int col  = idx & 15;
        const int b    = mtile * 16 + brow;
        const int n    = ncol0 + col;

        float z1 = zbuf[0][brow][col] + bias[n];
        float z2 = zbuf[1][brow][col] + bias[1024 + n];
        float z3 = zbuf[2][brow][col] + bias[2048 + n];
        float zo = zbuf[3][brow][col] + bias[3072 + n];
        if (zx) {
            const size_t zb = (size_t)b * 4096 + n;
            z1 += bf2f(zx[zb]);
            z2 += bf2f(zx[zb + 1024]);
            z3 += bf2f(zx[zb + 2048]);
            zo += bf2f(zx[zb + 3072]);
        }

        const float s1 = 1.f / (1.f + __expf(-z1));
        const float s2 = 1.f / (1.f + __expf(-z2));
        const float s3 = 1.f / (1.f + __expf(-z3));
        const float ot = tanhf(zo);

        const size_t ci = (size_t)b * 1024 + n;
        const float cn = c[ci] * s1 + ot * s2;
        c[ci] = cn;
        const float hn = tanhf(cn) * s3;
        hnext[ci] = f2bf_rne(hn);
        if (out) out[ci] = hn;
    }
}

// ---------- host ----------
extern "C" void kernel_launch(void* const* d_in, const int* in_sizes, int n_in,
                              void* d_out, int out_size, void* d_ws, size_t ws_size,
                              hipStream_t stream) {
    (void)in_sizes; (void)n_in; (void)out_size;

    const int T = 512, B = 32, NIN = 512, NOUT = 1024;
    const int CH = 64, NCHUNK = T / CH;          // 64-step chunks
    const int HROW = B * NOUT;                   // 32768 elems per h row

    const float* x  = (const float*)d_in[0];
    const float* h0 = (const float*)d_in[1];
    const float* c0 = (const float*)d_in[2];
    const float* W0 = (const float*)d_in[3];
    const float* b0 = (const float*)d_in[4];
    const float* W1 = (const float*)d_in[5];
    const float* b1 = (const float*)d_in[6];
    float* out = (float*)d_out;

    // ---- workspace layout (bytes, 256-aligned) ----
    char* ws = (char*)d_ws;
    unsigned short* x_bf  = (unsigned short*)(ws);                   //  0        : 16,777,216
    unsigned short* w0bf  = (unsigned short*)(ws + 16777216);        //            : 12,582,912
    unsigned short* w1bf  = (unsigned short*)(ws + 29360128);        //            : 16,777,216
    float*          cst   = (float*)         (ws + 46137344);        //            :    262,144
    unsigned short* hbf   = (unsigned short*)(ws + 46399488);        // fallback   :    262,144
    unsigned short* h0seq = (unsigned short*)(ws + 46661632);        // big: (CH+1)*64KB = 4,259,840
    unsigned short* h1seq = (unsigned short*)(ws + 50921472);        // big:               4,259,840
    unsigned short* zxbuf = (unsigned short*)(ws + 55181312);        // big: 2048*4096*2 = 16,777,216
    const size_t NEED_BIG = 71958528;

    const int nx  = T * B * NIN;
    const int nw0 = 4 * NOUT * (NIN + NOUT);
    const int nw1 = 4 * NOUT * (NOUT + NOUT);
    const int nst = 2 * B * NOUT;

    // ---- common prep: bf16 conversion, states, (h0,c0) -> out tail ----
    k_f2bf<<<(nx  + 255) / 256, 256, 0, stream>>>(x,  x_bf, nx);
    k_f2bf<<<(nw0 + 255) / 256, 256, 0, stream>>>(W0, w0bf, nw0);
    k_f2bf<<<(nw1 + 255) / 256, 256, 0, stream>>>(W1, w1bf, nw1);
    k_copyf<<<(nst + 255) / 256, 256, 0, stream>>>(c0, cst, nst);
    k_copyf<<<(nst + 255) / 256, 256, 0, stream>>>(h0, out + (size_t)T * HROW / 1, nst);
    k_copyf<<<(nst + 255) / 256, 256, 0, stream>>>(c0, out + (size_t)T * HROW + nst, nst);

    const dim3 sgrid(NOUT / 16, 2);

    if (ws_size >= NEED_BIG) {
        // ==== precomputed-input scheme: big parallel GEMMs + K=1024 recurrent steps ====
        k_f2bf<<<(HROW + 255) / 256, 256, 0, stream>>>(h0,        h0seq, HROW); // carry row 0
        k_f2bf<<<(HROW + 255) / 256, 256, 0, stream>>>(h0 + HROW, h1seq, HROW);

        for (int cchunk = 0; cchunk < NCHUNK; ++cchunk) {
            const int t0 = cchunk * CH;
            // layer 0: Zx = x_chunk @ Wx0^T  (M=2048, K1=512)
            gemm_zx<<<4096, 128, 0, stream>>>(x_bf + (size_t)t0 * B * NIN, NIN,
                                              w0bf, NIN + NOUT, zxbuf);
            for (int s = 0; s < CH; ++s)
                lstm_step<<<sgrid, 128, 0, stream>>>(
                    nullptr, NIN, zxbuf + (size_t)s * B * 4096,
                    h0seq + (size_t)s * HROW, w0bf, NIN + NOUT, b0,
                    cst, h0seq + (size_t)(s + 1) * HROW, nullptr);
            // layer 1: Zx = h0_chunk @ Wx1^T  (M=2048, K1=1024)
            gemm_zx<<<4096, 128, 0, stream>>>(h0seq + HROW, NOUT,
                                              w1bf, 2 * NOUT, zxbuf);
            for (int s = 0; s < CH; ++s)
                lstm_step<<<sgrid, 128, 0, stream>>>(
                    nullptr, NOUT, zxbuf + (size_t)s * B * 4096,
                    h1seq + (size_t)s * HROW, w1bf, 2 * NOUT, b1,
                    cst + HROW, h1seq + (size_t)(s + 1) * HROW,
                    out + (size_t)(t0 + s) * HROW);
            // carry last row -> row 0 for next chunk
            k_copybf<<<(HROW + 255) / 256, 256, 0, stream>>>(h0seq + (size_t)CH * HROW, h0seq, HROW);
            k_copybf<<<(HROW + 255) / 256, 256, 0, stream>>>(h1seq + (size_t)CH * HROW, h1seq, HROW);
        }
    } else {
        // ==== fallback: fused full-K step, double-buffered h ====
        auto HBF = [&](int layer, int buf) { return hbf + (size_t)(layer * 2 + buf) * HROW; };
        k_f2bf<<<(HROW + 255) / 256, 256, 0, stream>>>(h0,        HBF(0, 0), HROW);
        k_f2bf<<<(HROW + 255) / 256, 256, 0, stream>>>(h0 + HROW, HBF(1, 0), HROW);

        for (int t = 0; t < T; ++t) {
            const int p = t & 1;
            lstm_step<<<sgrid, 128, 0, stream>>>(
                x_bf + (size_t)t * B * NIN, NIN, nullptr,
                HBF(0, p), w0bf, NIN + NOUT, b0,
                cst, HBF(0, 1 - p), nullptr);
            lstm_step<<<sgrid, 128, 0, stream>>>(
                HBF(0, 1 - p), NOUT, nullptr,
                HBF(1, p), w1bf, 2 * NOUT, b1,
                cst + HROW, HBF(1, 1 - p),
                out + (size_t)t * HROW);
        }
    }
}
